// RetinaNet_48936857371070
// MI455X (gfx1250) — matrix operand order repaired and checked by reading, not compiled, not run on hardware
//
#include <hip/hip_runtime.h>
#include <hip/hip_bf16.h>

// ---------------------------------------------------------------------------
// RetinaNet head on MI455X (gfx1250): implicit-GEMM conv3x3 via
// V_WMMA_F32_16X16X32_BF16, f32 accumulation, 2x2 register tiling.
//
// ~580 GFLOP vs ~350MB unique traffic -> compute bound -> matrix pipes.
// - Activations converted once per level to NHWC bf16.
// - Weights pre-packed into exact per-lane WMMA B-fragment order.
// - Out-of-image taps read a zeroed workspace page (branchless, EXEC all-1s).
// - OT/O templated so all 8 loads per K-step use immediate offsets.
// - Each wave computes 32 px x 32 ch: 2 A-frags + 2 B-frags -> 4 WMMAs
//   (2 b128 loads per WMMA instead of 4).
// ---------------------------------------------------------------------------

typedef __attribute__((ext_vector_type(16))) __bf16 v16bf;
typedef __attribute__((ext_vector_type(8)))  __bf16 v8bf;
typedef __attribute__((ext_vector_type(8)))  float  v8f;

union Frag16 {
    v16bf v;
    struct { v8bf lo; v8bf hi; } p;
};

__device__ __forceinline__ __bf16 f2bf(float f) {
    unsigned u = __builtin_bit_cast(unsigned, f);
    unsigned r = u + 0x7FFFu + ((u >> 16) & 1u);      // round-to-nearest-even
    unsigned short h = (unsigned short)(r >> 16);
    return __builtin_bit_cast(__bf16, h);
}

// ---------------------------------------------------------------------------
// NCHW f32 -> NHWC bf16   (C == 256)
// ---------------------------------------------------------------------------
__global__ void nchw2nhwc_kernel(const float* __restrict__ in,
                                 __bf16* __restrict__ out,
                                 int HW, long total) {
    long idx = (long)blockIdx.x * blockDim.x + threadIdx.x;
    if (idx >= total) return;
    int  c    = (int)(idx & 255);
    long rest = idx >> 8;            // n*HW + p
    long n    = rest / HW;
    long p    = rest - n * HW;
    out[idx] = f2bf(in[(n * 256 + c) * HW + p]);
}

// ---------------------------------------------------------------------------
// Pack w[O,256,3,3] f32 -> bf16 WMMA B fragments.
// Fragment index = (tap*8 + kt)*OT + otile ; fragment = 32 lanes x 16 bf16
// (lane-contiguous 32B).  B (32x16, 16-bit): N=lane%16, h=lane/16:
// K=8h+e (e<8), K=16+8h+(e-8) (e>=8), K = channel offset within chunk kt.
// ---------------------------------------------------------------------------
__global__ void pack_w_kernel(const float* __restrict__ w,
                              __bf16* __restrict__ out,
                              int O, int OT) {
    long idx = (long)blockIdx.x * blockDim.x + threadIdx.x;
    long total = (long)72 * OT * 512;
    if (idx >= total) return;
    int  e     = (int)(idx & 15);
    int  lane  = (int)((idx >> 4) & 31);
    long frag  = idx >> 9;
    int  otile = (int)(frag % OT);
    int  tk    = (int)(frag / OT);
    int  kt    = tk & 7;
    int  tap   = tk >> 3;
    int  o     = otile * 16 + (lane & 15);
    int  c     = kt * 32 + ((lane & 16) ? 8 : 0) + ((e < 8) ? e : (8 + e));
    float val  = 0.f;
    if (o < O) val = w[((long)o * 256 + c) * 9 + tap];
    out[idx] = f2bf(val);
}

// ---------------------------------------------------------------------------
// conv3x3 (SAME, stride 1, C_in=256) as implicit GEMM with WMMA.
// Block: 256 threads = 8 waves = 4 M-quads x 2 N-quads
//        => 128 pixels x 64 out-channels per block.
// Wave tile: 32 px x 32 ch = 4 accumulators; K loop: 9 taps x 8 chunks,
// each K-step = 8 b128 loads + 4 WMMA(16x16x32).
// ---------------------------------------------------------------------------
template <int OT, int O, bool FINAL>
__global__ __launch_bounds__(256)
void conv3x3_wmma_kernel(const __bf16* __restrict__ in,
                         const __bf16* __restrict__ wpk,
                         const float*  __restrict__ bias,
                         const __bf16* __restrict__ zpage,
                         __bf16* __restrict__ out_bf,
                         float*  __restrict__ out_f,
                         int wshift, int hwshift, long nstride) {
    const int tid  = threadIdx.x;
    const int lane = tid & 31;
    const int wave = tid >> 5;
    const int mq   = wave & 3;                 // M quad within block
    const int nq   = wave >> 2;                // N quad within block
    const int ob   = blockIdx.y * 4 + nq * 2;  // first otile of this wave
    const int p0   = blockIdx.x * 128 + mq * 32;

    const int  W     = 1 << wshift;
    const int  H     = 1 << (hwshift - wshift);
    const int  mrow  = lane & 15;
    const bool hiK   = lane >= 16;
    const int  hioff = hiK ? 8 : 0;

    // two pixel rows handled by this wave's A fragments
    const int p_a = p0 + mrow;
    const int p_b = p_a + 16;
    const int n_a = p_a >> hwshift, n_b = p_b >> hwshift;
    const int pixmask = (1 << hwshift) - 1;
    const int y_a = (p_a & pixmask) >> wshift, y_b = (p_b & pixmask) >> wshift;
    const int x_a = p_a & (W - 1),             x_b = p_b & (W - 1);

    v8f acc00 = {}, acc01 = {}, acc10 = {}, acc11 = {};

    const __bf16* bptr = wpk + (long)ob * 512 + lane * 16;   // B base (2 otiles adjacent)
    const __bf16* zp   = zpage + hioff;

#pragma unroll
    for (int tap = 0; tap < 9; ++tap) {
        const int dy = tap / 3 - 1;
        const int dx = tap % 3 - 1;

        const int yya = y_a + dy, xxa = x_a + dx;
        const int yyb = y_b + dy, xxb = x_b + dx;
        const bool va = ((unsigned)yya < (unsigned)H) & ((unsigned)xxa < (unsigned)W);
        const bool vb = ((unsigned)yyb < (unsigned)H) & ((unsigned)xxb < (unsigned)W);
        const __bf16* abaseA =
            va ? (in + ((((long)n_a << hwshift) + ((long)yya << wshift) + xxa) << 8) + hioff)
               : zp;
        const __bf16* abaseB =
            vb ? (in + ((((long)n_b << hwshift) + ((long)yyb << wshift) + xxb) << 8) + hioff)
               : zp;
#pragma unroll
        for (int kt = 0; kt < 8; ++kt) {
            Frag16 a0, a1, b0, b1;
            const v8bf* apA = (const v8bf*)abaseA + kt * 4;      // kt*32 halves
            const v8bf* apB = (const v8bf*)abaseB + kt * 4;
            a0.p.lo = apA[0];  a0.p.hi = apA[2];
            a1.p.lo = apB[0];  a1.p.hi = apB[2];
            const v8bf* bp = (const v8bf*)bptr + (tap * 8 + kt) * OT * 64;
            b0.p.lo = bp[0];   b0.p.hi = bp[1];                  // otile ob
            b1.p.lo = bp[64];  b1.p.hi = bp[65];                 // otile ob+1 (+1024B)
            acc00 = __builtin_amdgcn_wmma_f32_16x16x32_bf16(
                false, a0.v, false, b0.v, (short)0, acc00, false, false);
            acc01 = __builtin_amdgcn_wmma_f32_16x16x32_bf16(
                false, a0.v, false, b1.v, (short)0, acc01, false, false);
            acc10 = __builtin_amdgcn_wmma_f32_16x16x32_bf16(
                false, a1.v, false, b0.v, (short)0, acc10, false, false);
            acc11 = __builtin_amdgcn_wmma_f32_16x16x32_bf16(
                false, a1.v, false, b1.v, (short)0, acc11, false, false);
        }
    }

    // C/D layout: lanes 0-15 -> N=lane, M=r ; lanes 16-31 -> N=lane-16, M=8+r
    const int   ocol0 = ob * 16 + (lane & 15);
    const int   ocol1 = ocol0 + 16;
    const int   mb0   = p0 + hioff;            // pixel base, A-frag 0
    const int   mb1   = mb0 + 16;              // pixel base, A-frag 1
    const float bv0   = (ocol0 < O) ? bias[ocol0] : 0.f;
    const float bv1   = (ocol1 < O) ? bias[ocol1] : 0.f;

#pragma unroll
    for (int r = 0; r < 8; ++r) {
        float v00 = acc00[r] + bv0;
        float v01 = acc01[r] + bv1;
        float v10 = acc10[r] + bv0;
        float v11 = acc11[r] + bv1;
        const int pm0 = mb0 + r;
        const int pm1 = mb1 + r;
        if (!FINAL) {
            v00 = v00 > 0.f ? v00 : 0.f;
            v01 = v01 > 0.f ? v01 : 0.f;
            v10 = v10 > 0.f ? v10 : 0.f;
            v11 = v11 > 0.f ? v11 : 0.f;
            out_bf[((long)pm0 << 8) + ocol0] = f2bf(v00);
            out_bf[((long)pm0 << 8) + ocol1] = f2bf(v01);
            out_bf[((long)pm1 << 8) + ocol0] = f2bf(v10);
            out_bf[((long)pm1 << 8) + ocol1] = f2bf(v11);
        } else {
            const int nn0 = pm0 >> hwshift, px0 = pm0 & pixmask;
            const int nn1 = pm1 >> hwshift, px1 = pm1 & pixmask;
            if (ocol0 < O) {
                out_f[(long)nn0 * nstride + (long)px0 * O + ocol0] = v00;
                out_f[(long)nn1 * nstride + (long)px1 * O + ocol0] = v10;
            }
            if (ocol1 < O) {
                out_f[(long)nn0 * nstride + (long)px0 * O + ocol1] = v01;
                out_f[(long)nn1 * nstride + (long)px1 * O + ocol1] = v11;
            }
        }
    }
}

// ---------------------------------------------------------------------------
extern "C" void kernel_launch(void* const* d_in, const int* in_sizes, int n_in,
                              void* d_out, int out_size, void* d_ws, size_t ws_size,
                              hipStream_t stream) {
    (void)in_sizes; (void)n_in; (void)out_size; (void)ws_size;

    const float* feats[5] = {(const float*)d_in[0], (const float*)d_in[1],
                             (const float*)d_in[2], (const float*)d_in[3],
                             (const float*)d_in[4]};
    const float* cls_conv_w = (const float*)d_in[5];
    const float* cls_conv_b = (const float*)d_in[6];
    const float* cls_out_w  = (const float*)d_in[7];
    const float* cls_out_b  = (const float*)d_in[8];
    const float* reg_conv_w = (const float*)d_in[9];
    const float* reg_conv_b = (const float*)d_in[10];
    const float* reg_out_w  = (const float*)d_in[11];
    const float* reg_out_b  = (const float*)d_in[12];

    // ---- workspace layout ----
    char* ws = (char*)d_ws;
    const size_t ACT = (size_t)2 * 16384 * 256 * 2;     // 16.78 MB (level-0 NHWC bf16)
    __bf16* bufIn = (__bf16*)(ws);
    __bf16* bufP  = (__bf16*)(ws + ACT);
    __bf16* bufQ  = (__bf16*)(ws + 2 * ACT);
    __bf16* wpk   = (__bf16*)(ws + 3 * ACT);
    const size_t WCONV = (size_t)72 * 16 * 512;         // O=256  (OT=16)
    const size_t WCLS  = (size_t)72 * 52 * 512;         // O=819  (OT=52)
    const size_t WREG  = (size_t)72 * 4  * 512;         // O=36   (OT padded to 4)
    __bf16* wpk_cls_conv = wpk;
    __bf16* wpk_cls_out  = wpk_cls_conv + 4 * WCONV;
    __bf16* wpk_reg_conv = wpk_cls_out + WCLS;
    __bf16* wpk_reg_out  = wpk_reg_conv + 4 * WCONV;
    __bf16* zpage        = wpk_reg_out + WREG;          // 2KB zero page

    hipMemsetAsync(zpage, 0, 2048, stream);

    // ---- pack all weights ----
    auto pack = [&](const float* w, __bf16* dst, int O, int OT) {
        long total = (long)72 * OT * 512;
        pack_w_kernel<<<dim3((unsigned)((total + 255) / 256)), dim3(256), 0, stream>>>(
            w, dst, O, OT);
    };
    for (int i = 0; i < 4; ++i)
        pack(cls_conv_w + (size_t)i * 256 * 256 * 9, wpk_cls_conv + i * WCONV, 256, 16);
    pack(cls_out_w, wpk_cls_out, 819, 52);
    for (int i = 0; i < 4; ++i)
        pack(reg_conv_w + (size_t)i * 256 * 256 * 9, wpk_reg_conv + i * WCONV, 256, 16);
    pack(reg_out_w, wpk_reg_out, 36, 4);

    // ---- per-level pipeline ----
    const int  sizes[5]  = {128, 64, 32, 16, 8};
    const int  shifts[5] = {7, 6, 5, 4, 3};             // log2(W)
    float* out_cls = (float*)d_out;
    float* out_reg = (float*)d_out + (size_t)2 * 196416 * 91;
    const long NS_CLS = (long)196416 * 91;
    const long NS_REG = (long)196416 * 4;
    long offA = 0;                                      // cumulative 9*sum(HW)

    for (int l = 0; l < 5; ++l) {
        const int s = sizes[l];
        const int hw = s * s;
        const int P = 2 * hw;                           // N*H*W, multiple of 128
        const int wsh  = shifts[l];
        const int hwsh = 2 * wsh;

        // NCHW f32 -> NHWC bf16
        long total = (long)P * 256;
        nchw2nhwc_kernel<<<dim3((unsigned)((total + 255) / 256)), dim3(256), 0, stream>>>(
            feats[l], bufIn, hw, total);

        const dim3 blk(256);
        const dim3 g256(P / 128, 4);                    // 256-out layers (OT=16)

        // ---- cls head ----
        conv3x3_wmma_kernel<16, 256, false><<<g256, blk, 0, stream>>>(
            bufIn, wpk_cls_conv + 0 * WCONV, cls_conv_b + 0 * 256, zpage, bufP, nullptr,
            wsh, hwsh, 0);
        conv3x3_wmma_kernel<16, 256, false><<<g256, blk, 0, stream>>>(
            bufP, wpk_cls_conv + 1 * WCONV, cls_conv_b + 1 * 256, zpage, bufQ, nullptr,
            wsh, hwsh, 0);
        conv3x3_wmma_kernel<16, 256, false><<<g256, blk, 0, stream>>>(
            bufQ, wpk_cls_conv + 2 * WCONV, cls_conv_b + 2 * 256, zpage, bufP, nullptr,
            wsh, hwsh, 0);
        conv3x3_wmma_kernel<16, 256, false><<<g256, blk, 0, stream>>>(
            bufP, wpk_cls_conv + 3 * WCONV, cls_conv_b + 3 * 256, zpage, bufQ, nullptr,
            wsh, hwsh, 0);
        conv3x3_wmma_kernel<52, 819, true><<<dim3(P / 128, 13), blk, 0, stream>>>(
            bufQ, wpk_cls_out, cls_out_b, zpage, nullptr, out_cls + offA * 91,
            wsh, hwsh, NS_CLS);

        // ---- reg head ----
        conv3x3_wmma_kernel<16, 256, false><<<g256, blk, 0, stream>>>(
            bufIn, wpk_reg_conv + 0 * WCONV, reg_conv_b + 0 * 256, zpage, bufP, nullptr,
            wsh, hwsh, 0);
        conv3x3_wmma_kernel<16, 256, false><<<g256, blk, 0, stream>>>(
            bufP, wpk_reg_conv + 1 * WCONV, reg_conv_b + 1 * 256, zpage, bufQ, nullptr,
            wsh, hwsh, 0);
        conv3x3_wmma_kernel<16, 256, false><<<g256, blk, 0, stream>>>(
            bufQ, wpk_reg_conv + 2 * WCONV, reg_conv_b + 2 * 256, zpage, bufP, nullptr,
            wsh, hwsh, 0);
        conv3x3_wmma_kernel<16, 256, false><<<g256, blk, 0, stream>>>(
            bufP, wpk_reg_conv + 3 * WCONV, reg_conv_b + 3 * 256, zpage, bufQ, nullptr,
            wsh, hwsh, 0);
        conv3x3_wmma_kernel<4, 36, true><<<dim3(P / 128, 1), blk, 0, stream>>>(
            bufQ, wpk_reg_out, reg_out_b, zpage, nullptr, out_reg + offA * 4,
            wsh, hwsh, NS_REG);

        offA += (long)9 * hw;
    }
}
